// FasterRCNN_90031104459493
// MI455X (gfx1250) — matrix operand order repaired and checked by reading, not compiled, not run on hardware
//
#include <hip/hip_runtime.h>
#include <hip/hip_bf16.h>

typedef __attribute__((ext_vector_type(16))) _Float16 v16h;
typedef __attribute__((ext_vector_type(8)))  float    v8f;
typedef unsigned int u32x4 __attribute__((ext_vector_type(4)));
typedef int          i32x8 __attribute__((ext_vector_type(8)));
typedef int          i32x4 __attribute__((ext_vector_type(4)));

static inline size_t wsAlign(size_t x) { return (x + 255) & ~(size_t)255; }

#if __has_builtin(__builtin_amdgcn_tensor_load_to_lds)
#define HAVE_TDM 1
#else
#define HAVE_TDM 0
#endif

// ---------------- fp32 -> fp16 convert ----------------
__global__ void k_f32_to_f16(const float* __restrict__ src, _Float16* __restrict__ dst, long n) {
  long i = (long)blockIdx.x * blockDim.x + threadIdx.x;
  if (i < n) dst[i] = (_Float16)src[i];
}

// ---------------- pack conv weights into WMMA A-fragment tiles ----------------
// A-frag (16x32 f16, wave32): lane<16 -> M=lane,    K in {0..7} U {16..23}
//                             lane>=16 -> M=lane-16, K in {8..15} U {24..31}
// M-tiles padded to groups of 4 (64 output channels) so LDS/TDM staging never
// goes out of bounds; padding rows are zero.
__global__ void k_pack_w(const float* __restrict__ w, _Float16* __restrict__ pk,
                         int COUT, int CIN, int KS) {
  const int K  = CIN * KS * KS;
  const int Kt = (K + 31) >> 5;
  const int Mt = ((COUT + 63) >> 6) << 2;     // padded tile count
  const long total = (long)Mt * Kt * 512;
  long idx = (long)blockIdx.x * blockDim.x + threadIdx.x;
  if (idx >= total) return;
  int e    = (int)(idx & 15);
  int lane = (int)((idx >> 4) & 31);
  long tile = idx >> 9;
  int kt = (int)(tile % Kt);
  int mt = (int)(tile / Kt);
  int v = e >> 1, pos = e & 1;
  int klocal = ((v < 4) ? 0 : 16) + ((lane < 16) ? 0 : 8) + (v & 3) * 2 + pos;
  int m = mt * 16 + (lane & 15);
  int k = kt * 32 + klocal;
  float val = 0.f;
  if (m < COUT && k < K) val = w[(long)m * K + k];
  pk[idx] = (_Float16)val;
}

// ---------------- implicit-GEMM conv via WMMA + LDS/TDM weight staging ----------
// Block = 128 threads = 4 wave32. Block computes 64 pixels x 64 output channels:
// wave w owns pixels p0 + w*16 .. +15 and accumulates 4 oc-tiles (4 WMMAs per
// K-chunk, one shared B gather). A-tiles (4 x 1KB) staged in LDS per K-chunk by
// the Tensor Data Mover (fallback: cooperative copy).
template <int KS>
__global__ __launch_bounds__(128)
void k_conv_wmma(const _Float16* __restrict__ in, const _Float16* __restrict__ wpk,
                 const float* __restrict__ bias, _Float16* __restrict__ out,
                 int CIN, int COUT, int H, int W, int relu) {
  constexpr int KK = KS * KS;
  const int HW  = H * W;
  const int K   = CIN * KK;
  const int Kt  = (K + 31) >> 5;
  constexpr int pad = KS >> 1;
  const int wave = threadIdx.x >> 5;
  const int lane = threadIdx.x & 31;
  const int mtBase = blockIdx.y * 4;
  const int p  = blockIdx.x * 64 + wave * 16 + (lane & 15); // HW % 64 == 0 always here
  const int y  = p / W;
  const int x  = p - y * W;
  // B-frag (32x16 f16): lane<16 holds K {0..7} U {16..23}, lane>=16 holds K {8..15} U {24..31}, N = lane%16
  const int khalf = (lane < 16) ? 0 : 8;

  __shared__ _Float16 shA[4 * 512];  // 4 A-tiles (sole LDS allocation -> offset 0)

  v8f acc0 = {}, acc1 = {}, acc2 = {}, acc3 = {};
  union BU { v16h v; _Float16 h[16]; };

  for (int kt = 0; kt < Kt; ++kt) {
    // ---- stage 4 A-tiles for this K-chunk into LDS ----
#if HAVE_TDM
    if (wave == 0) {
      // Tensor DMA descriptor (D#), 2D tile: 512 x 4 elems of 2B,
      // row stride = Kt*512 elems. See CDNA5 ISA 8.3/8.4.
      unsigned long long ga =
          (unsigned long long)(uintptr_t)(wpk + ((long)mtBase * Kt + kt) * 512);
      unsigned td0 = (unsigned)(Kt * 512);
      unsigned long long st = (unsigned long long)Kt * 512ull;
      u32x4 g0;
      g0[0] = 1u;                                   // count=1, user mode
      g0[1] = 0u;                                   // lds_addr = offset of shA
      g0[2] = (unsigned)(ga & 0xffffffffull);       // global_addr[31:0]
      g0[3] = (unsigned)((ga >> 32) & 0x01ffffffull) | (2u << 30); // addr hi + type=2
      i32x8 g1;
      g1[0] = (int)(1u << 16);                      // data_size = 2 bytes
      g1[1] = (int)((td0 & 0xffffu) << 16);         // tensor_dim0 low16
      g1[2] = (int)((td0 >> 16) & 0xffffu);         // tensor_dim0 hi16, dim1 lo16=0
      g1[3] = (int)(16u | (512u << 16));            // tensor_dim1 = 1<<20, tile_dim0=512
      g1[4] = 4;                                    // tile_dim1 = 4 rows
      g1[5] = (int)(unsigned)(st & 0xffffffffull);  // dim0_stride low32
      g1[6] = (int)(unsigned)((st >> 32) & 0xffffull); // dim0_stride hi16
      g1[7] = 0;
      i32x4 gz = {0, 0, 0, 0};
      i32x8 gz8 = {0, 0, 0, 0, 0, 0, 0, 0};
      __builtin_amdgcn_tensor_load_to_lds(g0, g1, gz, gz, gz8, 0);
    }
#else
    {
      const v16h* src = (const v16h*)(wpk + ((long)(mtBase + wave) * Kt + kt) * 512 +
                                      lane * 16);
      *(v16h*)&shA[wave * 512 + lane * 16] = *src;
    }
#endif

    // ---- im2col gather of this wave's B fragment (overlaps the DMA) ----
    BU b;
    const int kbase = kt * 32 + khalf;
#pragma unroll
    for (int e = 0; e < 16; ++e) {
      const int k = kbase + ((e < 8) ? e : e + 8);
      _Float16 vv = (_Float16)0.f;
      if (k < K) {
        const int cin = k / KK;            // KK is constexpr -> mul/shift
        const int r   = k - cin * KK;
        const int ky  = r / KS;
        const int kx  = r - ky * KS;
        const int yy  = y + ky - pad;
        const int xx  = x + kx - pad;
        if (yy >= 0 && yy < H && xx >= 0 && xx < W)
          vv = in[(long)cin * HW + (long)yy * W + xx];
      }
      b.h[e] = vv;
    }

#if HAVE_TDM
    if (wave == 0) __builtin_amdgcn_s_wait_tensorcnt(0);
#endif
    __syncthreads();    // A-tiles visible in LDS

    // ---- 4 WMMAs sharing one B fragment ----
    v16h a0 = *(const v16h*)&shA[0 * 512 + lane * 16];
    v16h a1 = *(const v16h*)&shA[1 * 512 + lane * 16];
    v16h a2 = *(const v16h*)&shA[2 * 512 + lane * 16];
    v16h a3 = *(const v16h*)&shA[3 * 512 + lane * 16];
    acc0 = __builtin_amdgcn_wmma_f32_16x16x32_f16(false, a0, false, b.v, (short)0, acc0, false, false);
    acc1 = __builtin_amdgcn_wmma_f32_16x16x32_f16(false, a1, false, b.v, (short)0, acc1, false, false);
    acc2 = __builtin_amdgcn_wmma_f32_16x16x32_f16(false, a2, false, b.v, (short)0, acc2, false, false);
    acc3 = __builtin_amdgcn_wmma_f32_16x16x32_f16(false, a3, false, b.v, (short)0, acc3, false, false);

    __syncthreads();    // all waves done with shA before next chunk overwrites
  }

  // C/D layout: VGPR r, lane<16 -> M = r, lane>=16 -> M = r+8; N = lane%16
  const int moff = ((lane < 16) ? 0 : 8);
#pragma unroll
  for (int r = 0; r < 8; ++r) {
    v8f* accs[4] = {&acc0, &acc1, &acc2, &acc3};
#pragma unroll
    for (int t = 0; t < 4; ++t) {
      const int oc = (mtBase + t) * 16 + moff + r;
      if (oc < COUT) {
        float vv = (*accs[t])[r] + bias[oc];
        if (relu) vv = fmaxf(vv, 0.f);
        out[(long)oc * HW + p] = (_Float16)vv;
      }
    }
  }
}

// ---------------- 2x2 maxpool ----------------
__global__ void k_maxpool2(const _Float16* __restrict__ in, _Float16* __restrict__ out,
                           int C, int H, int W) {
  const int Ho = H >> 1, Wo = W >> 1;
  long idx = (long)blockIdx.x * blockDim.x + threadIdx.x;
  long total = (long)C * Ho * Wo;
  if (idx >= total) return;
  int xo = (int)(idx % Wo);
  long t = idx / Wo;
  int yo = (int)(t % Ho);
  int c  = (int)(t / Ho);
  const _Float16* pp = in + (long)c * H * W + (long)(2 * yo) * W + 2 * xo;
  float m = fmaxf(fmaxf((float)pp[0], (float)pp[1]),
                  fmaxf((float)pp[W], (float)pp[W + 1]));
  out[idx] = (_Float16)m;
}

// ---------------- head post: proposals transpose + score softmax ----------------
__global__ void k_head_post(const _Float16* __restrict__ score, const _Float16* __restrict__ bbxo,
                            float* __restrict__ proposals, float* __restrict__ probs,
                            int Hf, int Wf) {
  const int nf = Hf * Wf;
  int idx = blockIdx.x * blockDim.x + threadIdx.x;
  if (idx >= nf * 9) return;
  int a = idx % 9;
  int f = idx / 9;
#pragma unroll
  for (int j = 0; j < 4; ++j)
    proposals[idx * 4 + j] = (float)bbxo[(long)(a * 4 + j) * nf + f];
  float s0 = (float)score[(long)(a * 2 + 0) * nf + f];
  float s1 = (float)score[(long)(a * 2 + 1) * nf + f];
  float m = fmaxf(s0, s1);
  float e0 = expf(s0 - m), e1 = expf(s1 - m);
  float d = e0 + e1;
  probs[idx * 2 + 0] = e0 / d;
  probs[idx * 2 + 1] = e1 / d;
}

// ---------------- anchor generation (faithful, incl. reference quirks) ----------------
__global__ void k_anchors(float* __restrict__ anchors, int Hf, int Wf,
                          float h_img, float w_img) {
  int idx = blockIdx.x * blockDim.x + threadIdx.x;
  if (idx >= Hf * Wf * 9) return;
  int a = idx % 9;
  int f = idx / 9;
  int ri = a / 3, si = a % 3;
  float ratio = (ri == 0) ? 0.5f : ((ri == 1) ? 1.0f : 2.0f);
  float scale = (si == 0) ? 8.0f : ((si == 1) ? 16.0f : 32.0f);
  float hs = h_img / (float)Hf;      // original divides BOTH strides by h_fmap
  float ws = h_img / (float)Hf;
  float b0 = 0.f, b1 = 0.f, b2 = ws - 1.f, b3 = hs - 1.f;
  float ya0 = b0 + 0.5f * (b3 - 1.f);
  float ya1 = b1 + 0.5f * (b2 - 1.f);
  float ya2 = b3 - b1 + 1.f;
  float ya3 = b2 - b0 + 1.f;
  float wr = rintf(sqrtf(ya2 * ya3 / ratio));
  float hr = rintf(wr * ratio);
  float a0 = ya0 - 0.5f * (wr - 1.f);
  float a1 = ya1 - 0.5f * (hr - 1.f);
  float a2 = ya0 + 0.5f * (wr - 1.f);
  float a3 = ya1 + 0.5f * (hr - 1.f);
  float y0 = a0 + 0.5f * (a3 - 1.f);
  float y1 = a1 + 0.5f * (a2 - 1.f);
  float y2 = a3 - a1 + 1.f;
  float y3 = a2 - a0 + 1.f;
  float w2 = y2 * scale, h2 = y3 * scale;
  float x0 = y0 - 0.5f * (w2 - 1.f), yy0 = y1 - 0.5f * (h2 - 1.f);
  float x1 = y0 + 0.5f * (w2 - 1.f), yy1 = y1 + 0.5f * (h2 - 1.f);
  int gi = f / Wf, gj = f % Wf;
  float sx = gj * ws, sy = gi * hs;
  x0 = fminf(fmaxf(x0 + sx, 0.f), w_img);
  x1 = fminf(fmaxf(x1 + sx, 0.f), w_img);
  yy0 = fminf(fmaxf(yy0 + sy, 0.f), h_img);
  yy1 = fminf(fmaxf(yy1 + sy, 0.f), h_img);
  anchors[idx * 4 + 0] = x0;
  anchors[idx * 4 + 1] = yy0;
  anchors[idx * 4 + 2] = x1;
  anchors[idx * 4 + 3] = yy1;
}

__global__ void k_zero(float* __restrict__ p, int n) {
  int i = blockIdx.x * blockDim.x + threadIdx.x;
  if (i < n) p[i] = 0.f;
}

// ---------------- IoU vs gt[0], targets, loss partial sums ----------------
__global__ void k_losses(const float* __restrict__ gt, const float* __restrict__ proposals,
                         const float* __restrict__ anchors, const float* __restrict__ probs,
                         float* __restrict__ tgt_out, float* __restrict__ accum, int n) {
  int i = blockIdx.x * blockDim.x + threadIdx.x;
  if (i >= n) return;
  float a0 = gt[0], a1 = gt[1], a2 = gt[2], a3 = gt[3];
  float b0 = anchors[i * 4 + 0], b1 = anchors[i * 4 + 1];
  float b2 = anchors[i * 4 + 2], b3 = anchors[i * 4 + 3];
  float areaA = (a2 - a0) * (a3 - a1);
  float areaB = (b2 - b0) * (b3 - b1);
  float lx = fmaxf(a0, b0), ly = fmaxf(a1, b1);
  float rx = fminf(a2, b2), ry = fminf(a3, b3);
  float iw = fmaxf(rx - lx, 0.f), ih = fmaxf(ry - ly, 0.f);
  float inter = iw * ih;
  float iou = inter / (areaA + areaB - inter);
  float tgt = (iou >= 0.7f) ? 1.f : ((iou <= 0.3f) ? 0.f : -1.f);
  tgt_out[i] = tgt;
  float npos = 0.f, sl1 = 0.f;
  if (tgt > 0.f) {
    npos = 1.f;
#pragma unroll
    for (int j = 0; j < 4; ++j) {
      float d = fabsf(proposals[i * 4 + j] - anchors[i * 4 + j]);
      sl1 += (d < 1.f) ? 0.5f * d * d : d - 0.5f;
    }
  }
  float p0 = probs[i * 2 + 0], p1 = probs[i * 2 + 1];
  float m = fmaxf(p0, p1);
  float lse = m + logf(expf(p0 - m) + expf(p1 - m));
  int t = (tgt > 0.f) ? 1 : 0;
  float nll = -(((t == 1) ? p1 : p0) - lse);
  float valid = (tgt != -1.f) ? 1.f : 0.f;
  atomicAdd(&accum[0], npos);
  atomicAdd(&accum[1], sl1);
  atomicAdd(&accum[2], nll * valid);
  atomicAdd(&accum[3], valid);
}

__global__ void k_finalize(const float* __restrict__ accum, float* __restrict__ out2) {
  out2[0] = accum[1] / fmaxf(accum[0] * 4.f, 1.f);
  out2[1] = accum[2] / fmaxf(accum[3], 1.f);
}

// =========================================================================
extern "C" void kernel_launch(void* const* d_in, const int* in_sizes, int n_in,
                              void* d_out, int out_size, void* d_ws, size_t ws_size,
                              hipStream_t stream) {
  (void)in_sizes; (void)n_in; (void)out_size; (void)ws_size;
  const float* image = (const float*)d_in[0];
  const float* gtbbx = (const float*)d_in[1];
  const float* cw[13]; const float* cb[13];
  for (int i = 0; i < 13; ++i) { cw[i] = (const float*)d_in[3 + 2 * i]; cb[i] = (const float*)d_in[4 + 2 * i]; }
  const float* rpn_w   = (const float*)d_in[29]; const float* rpn_b   = (const float*)d_in[30];
  const float* score_w = (const float*)d_in[31]; const float* score_b = (const float*)d_in[32];
  const float* bbx_w   = (const float*)d_in[33]; const float* bbx_b   = (const float*)d_in[34];

  // ---------------- workspace carve-up ----------------
  char* ws = (char*)d_ws;
  size_t off = 0;
  float* accum = (float*)(ws + off);            off = wsAlign(off + 4 * sizeof(float));
  _Float16* actA = (_Float16*)(ws + off);       off = wsAlign(off + (size_t)64 * 640 * 640 * sizeof(_Float16));
  _Float16* actB = (_Float16*)(ws + off);       off = wsAlign(off + (size_t)64 * 640 * 640 * sizeof(_Float16));
  _Float16* scoreBuf = (_Float16*)(ws + off);   off = wsAlign(off + (size_t)18 * 1600 * sizeof(_Float16));
  _Float16* bbxBuf = (_Float16*)(ws + off);     off = wsAlign(off + (size_t)36 * 1600 * sizeof(_Float16));
  float* probs = (float*)(ws + off);            off = wsAlign(off + (size_t)14400 * 2 * sizeof(float));

  struct Spec { const float* w; const float* b; int cin, cout, ks; };
  Spec specs[16] = {
    {cw[0],  cb[0],    3,  64, 3}, {cw[1],  cb[1],   64,  64, 3},
    {cw[2],  cb[2],   64, 128, 3}, {cw[3],  cb[3],  128, 128, 3},
    {cw[4],  cb[4],  128, 256, 3}, {cw[5],  cb[5],  256, 256, 3}, {cw[6], cb[6], 256, 256, 3},
    {cw[7],  cb[7],  256, 512, 3}, {cw[8],  cb[8],  512, 512, 3}, {cw[9], cb[9], 512, 512, 3},
    {cw[10], cb[10], 512, 512, 3}, {cw[11], cb[11], 512, 512, 3}, {cw[12], cb[12], 512, 512, 3},
    {rpn_w,  rpn_b,  512, 512, 3}, {score_w, score_b, 512, 18, 1}, {bbx_w, bbx_b, 512, 36, 1}
  };
  _Float16* pk[16];
  for (int i = 0; i < 16; ++i) {
    int MtPad = ((specs[i].cout + 63) / 64) * 4;
    size_t elems = (size_t)MtPad *
                   (size_t)((specs[i].cin * specs[i].ks * specs[i].ks + 31) / 32) * 512;
    pk[i] = (_Float16*)(ws + off);
    off = wsAlign(off + elems * sizeof(_Float16));
    long blk = (long)((elems + 255) / 256);
    k_pack_w<<<dim3((unsigned)blk), 256, 0, stream>>>(specs[i].w, pk[i],
                                                      specs[i].cout, specs[i].cin, specs[i].ks);
  }

  // ---------------- convert image ----------------
  {
    long n = (long)3 * 640 * 640;
    k_f32_to_f16<<<dim3((unsigned)((n + 255) / 256)), 256, 0, stream>>>(image, actA, n);
  }

  auto conv = [&](const _Float16* in, int li, _Float16* out, int H, int W, int relu) {
    const Spec& s = specs[li];
    dim3 g((unsigned)(H * W / 64), (unsigned)((s.cout + 63) / 64));
    if (s.ks == 3)
      k_conv_wmma<3><<<g, 128, 0, stream>>>(in, pk[li], s.b, out, s.cin, s.cout, H, W, relu);
    else
      k_conv_wmma<1><<<g, 128, 0, stream>>>(in, pk[li], s.b, out, s.cin, s.cout, H, W, relu);
  };
  auto pool = [&](const _Float16* in, _Float16* out, int C, int H, int W) {
    long total = (long)C * (H / 2) * (W / 2);
    k_maxpool2<<<dim3((unsigned)((total + 255) / 256)), 256, 0, stream>>>(in, out, C, H, W);
  };

  int H = 640, W = 640;
  conv(actA, 0, actB, H, W, 1);            // 3 -> 64
  conv(actB, 1, actA, H, W, 1);            // 64 -> 64
  pool(actA, actB, 64, H, W);  H >>= 1; W >>= 1;   // 320
  conv(actB, 2, actA, H, W, 1);            // 64 -> 128
  conv(actA, 3, actB, H, W, 1);            // 128 -> 128
  pool(actB, actA, 128, H, W); H >>= 1; W >>= 1;   // 160
  conv(actA, 4, actB, H, W, 1);            // 128 -> 256
  conv(actB, 5, actA, H, W, 1);
  conv(actA, 6, actB, H, W, 1);
  pool(actB, actA, 256, H, W); H >>= 1; W >>= 1;   // 80
  conv(actA, 7, actB, H, W, 1);            // 256 -> 512
  conv(actB, 8, actA, H, W, 1);
  conv(actA, 9, actB, H, W, 1);
  pool(actB, actA, 512, H, W); H >>= 1; W >>= 1;   // 40
  conv(actA, 10, actB, H, W, 1);
  conv(actB, 11, actA, H, W, 1);
  conv(actA, 12, actB, H, W, 1);
  conv(actB, 13, actA, H, W, 1);           // RPN conv + relu
  conv(actA, 14, scoreBuf, H, W, 0);       // score 1x1 (18ch)
  conv(actA, 15, bbxBuf,  H, W, 0);        // bbx 1x1 (36ch)

  // ---------------- outputs ----------------
  float* out_proposals = (float*)d_out;            // 14400*4
  float* out_anchors   = out_proposals + 57600;    // 14400*4
  float* out_targets   = out_anchors + 57600;      // 14400
  float* out_losses    = out_targets + 14400;      // 2

  k_head_post<<<dim3(57), 256, 0, stream>>>(scoreBuf, bbxBuf, out_proposals, probs, 40, 40);
  k_anchors<<<dim3(57), 256, 0, stream>>>(out_anchors, 40, 40, 640.f, 640.f);
  k_zero<<<dim3(1), 32, 0, stream>>>(accum, 4);
  k_losses<<<dim3(57), 256, 0, stream>>>(gtbbx, out_proposals, out_anchors, probs,
                                         out_targets, accum, 14400);
  k_finalize<<<dim3(1), 1, 0, stream>>>(accum, out_losses);
}
